// MultiHeadAttention_19387482374154
// MI455X (gfx1250) — compile-verified, matrix operand order checked by reading, not run on hardware
//
#include <hip/hip_runtime.h>

// Problem constants (from reference)
constexpr int BB  = 4;
constexpr int SS  = 2048;
constexpr int DD  = 1024;
constexpr int HH  = 8;
constexpr int DEP = 128;               // DD / HH
constexpr float QK_SCALE = 0.08838834764831845f;  // 1/sqrt(DEP)
constexpr float LOG2E    = 1.4426950408889634f;

typedef __attribute__((ext_vector_type(16))) _Float16 v16h;
typedef __attribute__((ext_vector_type(8)))  _Float16 v8h;
typedef __attribute__((ext_vector_type(4)))  _Float16 v4h;
typedef __attribute__((ext_vector_type(8)))  float    v8f;
typedef __attribute__((ext_vector_type(4)))  float    v4f;
typedef __attribute__((ext_vector_type(4)))  int      v4i;

// gfx1250 async global->LDS copy path (guarded; falls back to sync copy)
#if defined(__AMDGCN__) && __has_builtin(__builtin_amdgcn_global_load_async_to_lds_b128) && __has_builtin(__builtin_amdgcn_s_wait_asynccnt)
#define HAVE_ASYNC 1
#else
#define HAVE_ASYNC 0
#endif

// A/B WMMA fragment: 16 halves per lane = two aligned 16B chunks
union AFrag { v16h v; v8h h[2]; };

static __device__ __forceinline__ v8f vzero8() {
    v8f z;
#pragma unroll
    for (int i = 0; i < 8; ++i) z[i] = 0.0f;
    return z;
}

static __device__ __forceinline__ v8f wmma_f16(const v16h& a, const v16h& b, const v8f& c) {
    return __builtin_amdgcn_wmma_f32_16x16x32_f16(
        /*neg_a=*/false, a, /*neg_b=*/false, b,
        /*c_mod=*/(short)0, c, /*reuse_a=*/false, /*reuse_b=*/false);
}

// 16-byte global -> LDS copy (async on gfx1250, sync fallback elsewhere)
static __device__ __forceinline__ void cp16(const _Float16* g, _Float16* l) {
#if HAVE_ASYNC
    __builtin_amdgcn_global_load_async_to_lds_b128(
        (__attribute__((address_space(1))) v4i*)g,
        (__attribute__((address_space(3))) v4i*)l, 0, 0);
#else
    *(v8h*)l = *(const v8h*)g;
#endif
}

static __device__ __forceinline__ void wait_async_16() {
#if HAVE_ASYNC
    __builtin_amdgcn_s_wait_asynccnt(16);
#endif
}
static __device__ __forceinline__ void wait_async_8() {
#if HAVE_ASYNC
    __builtin_amdgcn_s_wait_asynccnt(8);
#endif
}
static __device__ __forceinline__ void wait_async_0() {
#if HAVE_ASYNC
    __builtin_amdgcn_s_wait_asynccnt(0);
#endif
}

// ---------------------------------------------------------------------------
// Projection: P = X @ W + bias, output as f16.
//   v_transpose == 0 : out[((b*H+h)*S + s)*DEP + d]   (row-major per head)
//   v_transpose == 1 : out[((b*H+h)*DEP + d)*S + s]   (V stored transposed)
// Block tile 128(M) x 64(N), 128 threads = 4 waves, each wave 64x32 via
// 4x2 v_wmma_f32_16x16x32_f16 tiles; K stepped by 32 through LDS f16 tiles.
// ---------------------------------------------------------------------------
__global__ __launch_bounds__(128)
void proj_kernel(const float* __restrict__ X, const float* __restrict__ W,
                 const float* __restrict__ bias, _Float16* __restrict__ out,
                 float out_scale, int v_transpose)
{
    __shared__ _Float16 ldsX[128 * 40];   // 128 rows x 32 K (80B rows, 16B aligned)
    __shared__ _Float16 ldsW[64 * 40];    // W^T tile: 64 cols x 32 K

    const int tid   = threadIdx.x;
    const int lane  = tid & 31;
    const int wave  = tid >> 5;
    const int waveM = wave >> 1;          // 0..1 -> 64-row half
    const int waveN = wave & 1;           // 0..1 -> 32-col half
    const int nl    = lane & 15;
    const int hi    = lane >> 4;
    const int blockM = blockIdx.y;        // 0..63  (8192/128)
    const int blockN = blockIdx.x;        // 0..15  (1024/64)

    v8f acc[4][2];
#pragma unroll
    for (int mt = 0; mt < 4; ++mt)
#pragma unroll
        for (int nt = 0; nt < 2; ++nt) acc[mt][nt] = vzero8();

    for (int k0 = 0; k0 < DD; k0 += 32) {
        // Stage X tile (128x32 f32 -> f16): one row per thread, 8x float4.
        {
            const float* src = X + ((size_t)blockM * 128 + tid) * DD + k0;
            _Float16* dst = &ldsX[tid * 40];
#pragma unroll
            for (int j = 0; j < 8; ++j) {
                v4f f = *(const v4f*)(src + j * 4);
                v4h hv;
#pragma unroll
                for (int e = 0; e < 4; ++e) hv[e] = (_Float16)f[e];
                *(v4h*)(dst + j * 4) = hv;
            }
        }
        // Stage W tile (32x64 f32) transposed into ldsW[n][k] as f16.
        {
            const int kr = tid & 31;       // K row within tile
            const int cg = tid >> 5;       // column group of 16
            const float* src = W + (size_t)(k0 + kr) * DD + blockN * 64 + cg * 16;
#pragma unroll
            for (int j = 0; j < 4; ++j) {
                v4f f = *(const v4f*)(src + j * 4);
#pragma unroll
                for (int e = 0; e < 4; ++e)
                    ldsW[(cg * 16 + j * 4 + e) * 40 + kr] = (_Float16)f[e];
            }
        }
        __syncthreads();

#pragma unroll
        for (int mt = 0; mt < 4; ++mt) {
            AFrag a;
            const _Float16* ap = &ldsX[(waveM * 64 + mt * 16 + nl) * 40];
            a.h[0] = *(const v8h*)(ap + hi * 8);
            a.h[1] = *(const v8h*)(ap + 16 + hi * 8);
#pragma unroll
            for (int nt = 0; nt < 2; ++nt) {
                AFrag b;
                const _Float16* bp = &ldsW[(waveN * 32 + nt * 16 + nl) * 40];
                b.h[0] = *(const v8h*)(bp + hi * 8);
                b.h[1] = *(const v8h*)(bp + 16 + hi * 8);
                acc[mt][nt] = wmma_f16(a.v, b.v, acc[mt][nt]);
            }
        }
        __syncthreads();
    }

    // Epilogue: bias, scale, convert to f16, scatter into head layout.
#pragma unroll
    for (int nt = 0; nt < 2; ++nt) {
        const int n_global = blockN * 64 + waveN * 32 + nt * 16 + nl;
        const float bn = bias[n_global];
        const int h = n_global >> 7;          // / DEP
        const int d = n_global & 127;         // % DEP
#pragma unroll
        for (int mt = 0; mt < 4; ++mt) {
#pragma unroll
            for (int r = 0; r < 8; ++r) {
                const int m_global = blockM * 128 + waveM * 64 + mt * 16 + r + hi * 8;
                const int b   = m_global >> 11;   // / SS
                const int s   = m_global & 2047;  // % SS
                const float val = (acc[mt][nt][r] + bn) * out_scale;
                size_t off;
                if (v_transpose) off = (((size_t)b * HH + h) * DEP + d) * SS + s;
                else             off = (((size_t)b * HH + h) * SS + s) * DEP + d;
                out[off] = (_Float16)val;
            }
        }
    }
}

// ---------------------------------------------------------------------------
// Flash attention over f16 Qh[B,H,S,DEP] (pre-scaled), Kh[B,H,S,DEP],
// Vt[B,H,DEP,S]; additive mask [B,S,S]; f32 output [B,S,D] head-concat.
// 128 threads = 4 waves; each wave owns 16 query rows (block = 64 q rows).
//
// K/V are staged into LDS once per block (shared by all 4 waves) with
// gfx1250 async global->LDS copies, double-buffered K for pipelining:
//   issue order:  K0 | V0,K1 | V1,K2 | ... (in-order ASYNC completion)
//   per iter:     wait<=16 (K cur ready) -> scores/softmax
//                 wait<=8  (V cur ready) -> P@V; last iter uses <=8 / <=0.
// ---------------------------------------------------------------------------
__global__ __launch_bounds__(128)
void attn_kernel(const _Float16* __restrict__ Qh, const _Float16* __restrict__ Kh,
                 const _Float16* __restrict__ Vt, const float* __restrict__ mask,
                 float* __restrict__ out)
{
    __shared__ _Float16 ldsK[2][64 * 136];  // 64 keys x 128 depth, 272B rows (double buf)
    __shared__ _Float16 ldsV[128 * 72];     // 128 depth-rows x 64 keys, 144B rows
    __shared__ _Float16 ldsP[4][16 * 72];   // per-wave 16x64 P tile, 144B rows

    const int tid  = threadIdx.x;
    const int lane = tid & 31;
    const int wave = tid >> 5;
    const int nl   = lane & 15;
    const int hi   = lane >> 4;

    const int bid   = blockIdx.x;          // BB*HH*(SS/64) = 1024 blocks
    const int qblk  = bid & 31;            // SS/64 = 32 query blocks
    const int bh    = bid >> 5;            // b*H + h
    const int b     = bh >> 3;
    const int h     = bh & 7;
    const int qbase = qblk * 64 + wave * 16;

    const _Float16* Qp    = Qh + ((size_t)bh * SS + qbase) * DEP;
    const _Float16* Kbase = Kh + (size_t)bh * SS * DEP;
    const _Float16* Vbase = Vt + (size_t)bh * DEP * SS;
    const float*    mrow  = mask + ((size_t)b * SS + qbase) * SS;

    // Q A-fragments: 4 K-steps of 32 over depth=128, resident in VGPRs.
    AFrag qf[4];
#pragma unroll
    for (int ks = 0; ks < 4; ++ks) {
        const _Float16* p = Qp + (size_t)nl * DEP + ks * 32;
        qf[ks].h[0] = *(const v8h*)(p + hi * 8);
        qf[ks].h[1] = *(const v8h*)(p + 16 + hi * 8);
    }

    v8f o[8];
#pragma unroll
    for (int dt = 0; dt < 8; ++dt) o[dt] = vzero8();
    float mi[8], li[8];
#pragma unroll
    for (int r = 0; r < 8; ++r) { mi[r] = -3.0e38f; li[r] = 0.0f; }

    _Float16* Pl = &ldsP[wave][0];

    // Prologue: stage K chunk 0 into buffer 0 (8 x 16B per thread).
#pragma unroll
    for (int j = 0; j < 8; ++j) {
        const int u = tid + j * 128;               // 0..1023
        const int r = u >> 4, c = (u & 15) * 8;    // key row, half-offset
        cp16(Kbase + (size_t)r * DEP + c, &ldsK[0][r * 136 + c]);
    }

    for (int t0 = 0; t0 < SS; t0 += 64) {
        const int  it       = t0 >> 6;
        const int  cur      = it & 1;
        const int  nxt      = cur ^ 1;
        const bool has_next = (t0 + 64) < SS;

        // Issue V(cur) then (optionally) K(next): 8+8 async copies per thread.
#pragma unroll
        for (int j = 0; j < 8; ++j) {
            const int u = tid + j * 128;
            const int r = u >> 3, c = (u & 7) * 8;  // depth row, key half-offset
            cp16(Vbase + (size_t)r * SS + t0 + c, &ldsV[r * 72 + c]);
        }
        if (has_next) {
#pragma unroll
            for (int j = 0; j < 8; ++j) {
                const int u = tid + j * 128;
                const int r = u >> 4, c = (u & 15) * 8;
                cp16(Kbase + (size_t)(t0 + 64 + r) * DEP + c, &ldsK[nxt][r * 136 + c]);
            }
            __builtin_prefetch(mrow + (size_t)nl * SS + t0 + 64, 0, 3);
            wait_async_16();                 // K(cur) landed; V(cur),K(next) in flight
        } else {
            wait_async_8();                  // K(cur) landed; V(cur) in flight
        }
        __syncthreads();                     // K(cur) visible to all waves

        // ---- scores = Qh @ Kh^T (Qh pre-scaled by 1/sqrt(DEP)) ----
        v8f sc[4];
#pragma unroll
        for (int nt = 0; nt < 4; ++nt) {
            v8f a = vzero8();
            const _Float16* kp = &ldsK[cur][(nt * 16 + nl) * 136];
#pragma unroll
            for (int ks = 0; ks < 4; ++ks) {
                AFrag bf;
                bf.h[0] = *(const v8h*)(kp + ks * 32 + hi * 8);
                bf.h[1] = *(const v8h*)(kp + ks * 32 + 16 + hi * 8);
                a = wmma_f16(qf[ks].v, bf.v, a);
            }
            sc[nt] = a;
        }

        // ---- additive mask ----
#pragma unroll
        for (int nt = 0; nt < 4; ++nt) {
#pragma unroll
            for (int r = 0; r < 8; ++r) {
                const float mv = mrow[(size_t)(r + hi * 8) * SS + t0 + nt * 16 + nl];
                sc[nt][r] += mv * -1.0e9f;
            }
        }

        // ---- online softmax (row stats via 16-lane butterfly) ----
#pragma unroll
        for (int r = 0; r < 8; ++r) {
            float v = fmaxf(fmaxf(sc[0][r], sc[1][r]), fmaxf(sc[2][r], sc[3][r]));
            v = fmaxf(v, __shfl_xor(v, 1, 32));
            v = fmaxf(v, __shfl_xor(v, 2, 32));
            v = fmaxf(v, __shfl_xor(v, 4, 32));
            v = fmaxf(v, __shfl_xor(v, 8, 32));
            const float mnew = fmaxf(v, mi[r]);

            float rs = 0.0f;
#pragma unroll
            for (int nt = 0; nt < 4; ++nt) {
                const float p = __builtin_amdgcn_exp2f((sc[nt][r] - mnew) * LOG2E);
                rs += p;
                Pl[(r + hi * 8) * 72 + nt * 16 + nl] = (_Float16)p;
            }
            rs += __shfl_xor(rs, 1, 32);
            rs += __shfl_xor(rs, 2, 32);
            rs += __shfl_xor(rs, 4, 32);
            rs += __shfl_xor(rs, 8, 32);

            const float alpha = __builtin_amdgcn_exp2f((mi[r] - mnew) * LOG2E);
            li[r] = li[r] * alpha + rs;
            mi[r] = mnew;
#pragma unroll
            for (int dt = 0; dt < 8; ++dt) o[dt][r] *= alpha;
        }

        if (has_next) wait_async_8();        // V(cur) landed; K(next) in flight
        else          wait_async_0();
        __syncthreads();                     // V(cur) visible to all waves

        // ---- O += P @ V ----
#pragma unroll
        for (int ks = 0; ks < 2; ++ks) {
            AFrag pf;
            const _Float16* pp = Pl + (size_t)nl * 72 + ks * 32;
            pf.h[0] = *(const v8h*)(pp + hi * 8);
            pf.h[1] = *(const v8h*)(pp + 16 + hi * 8);
#pragma unroll
            for (int dt = 0; dt < 8; ++dt) {
                AFrag vf;
                const _Float16* vp = &ldsV[(dt * 16 + nl) * 72] + ks * 32;
                vf.h[0] = *(const v8h*)(vp + hi * 8);
                vf.h[1] = *(const v8h*)(vp + 16 + hi * 8);
                o[dt] = wmma_f16(pf.v, vf.v, o[dt]);
            }
        }

        __syncthreads();                     // all reads of K(cur)/V done before overwrite
    }

    // ---- normalize and write head-concat output ----
#pragma unroll
    for (int r = 0; r < 8; ++r) {
        const float inv = 1.0f / li[r];
        const int qrow = qbase + r + hi * 8;
        float* orow = out + ((size_t)b * SS + qrow) * DD + h * DEP;
#pragma unroll
        for (int dt = 0; dt < 8; ++dt)
            orow[dt * 16 + nl] = o[dt][r] * inv;
    }
}

// ---------------------------------------------------------------------------
extern "C" void kernel_launch(void* const* d_in, const int* in_sizes, int n_in,
                              void* d_out, int out_size, void* d_ws, size_t ws_size,
                              hipStream_t stream)
{
    const float* q    = (const float*)d_in[0];
    const float* k    = (const float*)d_in[1];
    const float* v    = (const float*)d_in[2];
    const float* mask = (const float*)d_in[3];
    const float* wq   = (const float*)d_in[4];
    const float* wqb  = (const float*)d_in[5];
    const float* wk   = (const float*)d_in[6];
    const float* wkb  = (const float*)d_in[7];
    const float* wv   = (const float*)d_in[8];
    const float* wvb  = (const float*)d_in[9];
    float* out = (float*)d_out;

    const size_t elems = (size_t)BB * SS * DD;       // 8M halves = 16 MB each
    _Float16* Qh = (_Float16*)d_ws;
    _Float16* Kh = Qh + elems;
    _Float16* Vt = Kh + elems;

    dim3 pgrid(DD / 64, (BB * SS) / 128);            // 16 x 64 blocks
    proj_kernel<<<pgrid, 128, 0, stream>>>(q, wq, wqb, Qh, QK_SCALE, 0);
    proj_kernel<<<pgrid, 128, 0, stream>>>(k, wk, wkb, Kh, 1.0f, 0);
    proj_kernel<<<pgrid, 128, 0, stream>>>(v, wv, wvb, Vt, 1.0f, 1);

    attn_kernel<<<BB * HH * (SS / 64), 128, 0, stream>>>(Qh, Kh, Vt, mask, out);
}
